// CNN2LWithRPE_45174466019882
// MI455X (gfx1250) — compile-verified, matrix operand order checked by reading, not run on hardware
//
#include <hip/hip_runtime.h>
#include <hip/hip_bf16.h>

// ---------------- model dims ----------------
#define BB   16
#define LL   2048
#define DD   128
#define HH   4
#define HD   32
#define FF   256
#define MD   32
#define C1   128
#define C2   256
#define KK   5
#define L2IN 1024     // after maxpool
#define L2OUT 1020    // conv2 valid outputs
#define EPS  1e-5f

typedef __attribute__((ext_vector_type(16))) __bf16 v16bf;
typedef __attribute__((ext_vector_type(8)))  __bf16 v8bf;
typedef __attribute__((ext_vector_type(8)))  float  v8f;

// -------------------------------------------------------------------------
// WMMA helpers.  D = A(16x32 bf16) x B(32x16 bf16) + C(16x16 f32)
// -------------------------------------------------------------------------
__device__ __forceinline__ v8f wmma_bf16(v16bf a, v16bf b, v8f c) {
  return __builtin_amdgcn_wmma_f32_16x16x32_bf16(
      false, a, false, b, (short)0, c, false, false);
}

// Load a 16x32 bf16 fragment from a row-major matrix (row stride = ld elems).
// ISA 16-bit A layout: lanes 0-15 hold row (lane), K chunks [0..7],[16..23];
// lanes 16-31 hold row (lane-16), K chunks [8..15],[24..31].
__device__ __forceinline__ v16bf load_frag(const __bf16* base, int ld) {
  const int lane = threadIdx.x & 31;
  const int row  = lane & 15;
  const int koff = (lane & 16) ? 8 : 0;
  union { v16bf v; v8bf h[2]; } u;
  u.h[0] = *(const v8bf*)(base + (size_t)row * ld + koff);
  u.h[1] = *(const v8bf*)(base + (size_t)row * ld + koff + 16);
  return u.v;
}

// prefetch the lane's row of the next K tile (emits global_prefetch_b8)
__device__ __forceinline__ void prefetch_frag(const __bf16* base, int ld) {
  const int lane = threadIdx.x & 31;
  __builtin_prefetch(base + (size_t)(lane & 15) * ld + ((lane & 16) ? 8 : 0), 0, 0);
}

// -------------------------------------------------------------------------
// 0) embedding: x[b][l][c] = c<120 ? emb[X]*sa : ptm_emb[ptm]   (f32 + bf16)
// -------------------------------------------------------------------------
__global__ __launch_bounds__(256)
void embed_kernel(const int* X, const float* sa, const int* ptm,
                  const float* emb, const float* pemb,
                  float* xf, __bf16* xb) {
  int i = blockIdx.x * 256 + threadIdx.x;          // B*L*128
  if (i >= BB * LL * DD) return;
  int c = i & 127, bl = i >> 7;
  float v;
  if (c < 120) v = emb[X[bl] * 120 + c] * sa[bl];
  else         v = pemb[ptm[bl] * 8 + (c - 120)];
  xf[i] = v;
  xb[i] = (__bf16)v;
}

// generic f32 -> bf16
__global__ __launch_bounds__(256)
void f2bf_kernel(const float* a, __bf16* o, int n) {
  int i = blockIdx.x * 256 + threadIdx.x;
  if (i < n) o[i] = (__bf16)a[i];
}

// conv weight transform: wt[t][co][ci] = w[co][ci][t]   (f32 -> bf16)
__global__ __launch_bounds__(256)
void convtr_kernel(const float* w, __bf16* wt, int Cout, int Cin) {
  int i = blockIdx.x * 256 + threadIdx.x;
  int n = Cout * Cin * KK;
  if (i >= n) return;
  int ci = i % Cin, co = (i / Cin) % Cout, t = i / (Cin * Cout);
  wt[i] = (__bf16)w[((size_t)co * Cin + ci) * KK + t];
}

// -------------------------------------------------------------------------
// 1) generic GEMM: C[M,N] = A[M,K] @ W[N,K]^T + bias (opt relu, f32/bf16 out)
//    2x2 register blocking: each wave owns a 32x32 tile (4 WMMA / 4 loads).
//    block = 8 waves (2x4) -> 64 rows x 128 cols.
// -------------------------------------------------------------------------
__global__ __launch_bounds__(256)
void gemm_bf16(const __bf16* __restrict__ A, const __bf16* __restrict__ W,
               const float* __restrict__ bias,
               float* Cf, __bf16* Cb, int M, int N, int K, int relu) {
  const int wave = threadIdx.x >> 5, lane = threadIdx.x & 31;
  const int m0 = (blockIdx.x * 2 + (wave >> 2)) * 32;
  const int n0 = (blockIdx.y * 4 + (wave & 3)) * 32;
  if (m0 >= M || n0 >= N) return;
  v8f c00 = {}, c01 = {}, c10 = {}, c11 = {};
  const __bf16* pa0 = A + (size_t)m0 * K;
  const __bf16* pa1 = A + (size_t)(m0 + 16) * K;
  const __bf16* pw0 = W + (size_t)n0 * K;
  const __bf16* pw1 = W + (size_t)(n0 + 16) * K;
  for (int k0 = 0; k0 < K; k0 += 32) {
    if (k0 + 32 < K) { prefetch_frag(pa0 + k0 + 32, K); prefetch_frag(pa1 + k0 + 32, K); }
    v16bf a0 = load_frag(pa0 + k0, K);
    v16bf a1 = load_frag(pa1 + k0, K);
    v16bf w0 = load_frag(pw0 + k0, K);
    v16bf w1 = load_frag(pw1 + k0, K);
    c00 = wmma_bf16(a0, w0, c00);
    c01 = wmma_bf16(a0, w1, c01);
    c10 = wmma_bf16(a1, w0, c10);
    c11 = wmma_bf16(a1, w1, c11);
  }
  const int rl = (lane >> 4) << 3, cn = lane & 15;
  const v8f accs[4] = {c00, c01, c10, c11};
  for (int q = 0; q < 4; q++) {
    const int mo = (q >> 1) * 16, col = n0 + (q & 1) * 16 + cn;
    const float bv = bias ? bias[col] : 0.f;
    for (int i = 0; i < 8; i++) {
      float v = accs[q][i] + bv;
      if (relu) v = fmaxf(v, 0.f);
      size_t idx = (size_t)(m0 + mo + rl + i) * N + col;
      if (Cf) Cf[idx] = v;
      if (Cb) Cb[idx] = (__bf16)v;
    }
  }
}

// V transpose: vt[((b*H+h)*HD+d)*L + m] = qkv[(b*L+m)*384 + 256 + h*HD + d]
__global__ __launch_bounds__(256)
void vtrans_kernel(const __bf16* qkv, __bf16* vt) {
  int i = blockIdx.x * 256 + threadIdx.x;          // 2^22 total
  if (i >= BB * HH * HD * LL) return;
  int m = i & (LL - 1);
  int d = (i >> 11) & 31;
  int h = (i >> 16) & 3;
  int b = i >> 18;
  vt[i] = qkv[((size_t)(b * LL + m)) * 384 + 256 + h * HD + d];
}

// -------------------------------------------------------------------------
// 2) fused attention: per (b,h,16-row tile) -> scores+bias in LDS, softmax,
//    PV with WMMA.  LDS = 128KB scores + 64KB probs + 16KB partials = 208KB.
// -------------------------------------------------------------------------
#define ATTN_SMEM (16*LL*4 + 16*LL*2 + 8*16*HD*4)

__global__ __launch_bounds__(256)
void attn_kernel(const __bf16* __restrict__ qkv, const __bf16* __restrict__ vt,
                 const float* __restrict__ rpe, __bf16* __restrict__ ob) {
  extern __shared__ char smem[];
  float*  sc = (float*)smem;                        // [16][2048]
  __bf16* pr = (__bf16*)(smem + 16 * LL * 4);       // [16][2048]
  float*  op = (float*)(smem + 16 * LL * 4 + 16 * LL * 2); // [8][16][32]

  const int wave = threadIdx.x >> 5, lane = threadIdx.x & 31;
  const int rt = blockIdx.x & 127;
  const int h  = (blockIdx.x >> 7) & 3;
  const int b  = blockIdx.x >> 9;
  const int q0 = rt * 16;
  const float scale = 0.17677669529663687f;         // 1/sqrt(32)
  const int rl = (lane >> 4) << 3, cn = lane & 15;

  // Q fragment (16 rows x HD=32), reused by all col tiles of this wave
  const v16bf qf = load_frag(qkv + ((size_t)(b * LL) + q0) * 384 + h * HD, 384);
  const __bf16* kbase = qkv + (size_t)(b * LL) * 384 + 128 + h * HD;

  // ---- scores = Q K^T * scale + rpe bias -> LDS ----
  for (int ct = wave; ct < LL / 16; ct += 8) {
    v16bf kf = load_frag(kbase + (size_t)(ct * 16) * 384, 384);
    v8f acc = {};
    acc = wmma_bf16(qf, kf, acc);
    for (int i = 0; i < 8; i++) {
      int r = rl + i, col = ct * 16 + cn;
      int rel = col - (q0 + r);
      rel = rel < -MD ? -MD : (rel > MD ? MD : rel);
      sc[r * LL + col] = acc[i] * scale + rpe[(rel + MD) * HH + h];
    }
  }
  __syncthreads();

  // ---- softmax over each of the 16 rows (16 lanes per row) ----
  {
    int rr = threadIdx.x >> 4, sb = threadIdx.x & 15;
    float mx = -3.4e38f;
    for (int j = sb; j < LL; j += 16) mx = fmaxf(mx, sc[rr * LL + j]);
    for (int o = 8; o; o >>= 1) mx = fmaxf(mx, __shfl_xor(mx, o));
    float sum = 0.f;
    for (int j = sb; j < LL; j += 16) {
      float e = __expf(sc[rr * LL + j] - mx);
      sc[rr * LL + j] = e;
      sum += e;
    }
    for (int o = 8; o; o >>= 1) sum += __shfl_xor(sum, o);
    float inv = 1.f / sum;
    for (int j = sb; j < LL; j += 16)
      pr[rr * LL + j] = (__bf16)(sc[rr * LL + j] * inv);
  }
  __syncthreads();

  // ---- PV: o[16,32] = probs[16,2048] @ V[2048,32]; K split over 8 waves,
  //      one probs fragment feeds both HD tiles (2 WMMA per A-load) ----
  const __bf16* vb = vt + ((size_t)(b * HH + h) * HD) * LL;
  {
    v8f acc0 = {}, acc1 = {};
    for (int kk = 0; kk < 8; ++kk) {
      int k0 = wave * 256 + kk * 32;
      v16bf af  = load_frag(pr + k0, LL);                     // probs (LDS)
      v16bf bf0 = load_frag(vb + k0, LL);                     // V^T rows 0..15
      v16bf bf1 = load_frag(vb + (size_t)16 * LL + k0, LL);   // V^T rows 16..31
      acc0 = wmma_bf16(af, bf0, acc0);
      acc1 = wmma_bf16(af, bf1, acc1);
    }
    for (int i = 0; i < 8; i++) {
      op[(wave * 16 + rl + i) * HD + cn]      = acc0[i];
      op[(wave * 16 + rl + i) * HD + 16 + cn] = acc1[i];
    }
  }
  __syncthreads();

  for (int e = threadIdx.x; e < 16 * HD; e += 256) {
    int rr = e >> 5, dd = e & 31;
    float s = 0.f;
    for (int w = 0; w < 8; ++w) s += op[(w * 16 + rr) * HD + dd];
    ob[((size_t)(b * LL) + q0 + rr) * DD + h * HD + dd] = (__bf16)s;
  }
}

// -------------------------------------------------------------------------
// 3) residual add + LayerNorm (one wave per 128-elem row)
// -------------------------------------------------------------------------
__global__ __launch_bounds__(256)
void add_ln_kernel(const float* __restrict__ a, const float* __restrict__ r,
                   const float* g, const float* bb,
                   float* outf, __bf16* outb, int rowsPerB, int rowOff) {
  const int wave = threadIdx.x >> 5, lane = threadIdx.x & 31;
  const int row = blockIdx.x * 8 + wave;
  if (row >= BB * LL) return;
  const float* pa = a + (size_t)row * DD;
  const float* pr = r + (size_t)row * DD;
  float v[4], s = 0.f, s2 = 0.f;
  for (int j = 0; j < 4; j++) {
    int c = lane * 4 + j;
    float t = pa[c] + pr[c];
    v[j] = t; s += t; s2 += t * t;
  }
  for (int o = 16; o; o >>= 1) { s += __shfl_xor(s, o); s2 += __shfl_xor(s2, o); }
  float mean = s * (1.f / DD);
  float var  = s2 * (1.f / DD) - mean * mean;
  float inv  = rsqrtf(var + EPS);
  int b = row / LL, l = row % LL;
  size_t orow = ((size_t)b * rowsPerB + l + rowOff) * DD;
  for (int j = 0; j < 4; j++) {
    int c = lane * 4 + j;
    float t = (v[j] - mean) * inv * g[c] + bb[c];
    if (outf) outf[(size_t)row * DD + c] = t;
    if (outb) outb[orow + c] = (__bf16)t;
  }
}

// zero the 2-row halo of xpad [B][2052][128]
__global__ __launch_bounds__(256)
void xpad_zero_kernel(__bf16* xpad) {
  int i = blockIdx.x * 256 + threadIdx.x;          // 16*4*128 = 8192
  if (i >= BB * 4 * DD) return;
  int c = i & 127, r = (i >> 7) & 3, b = i >> 9;
  int row = (r < 2) ? r : (2050 + r - 2);
  xpad[((size_t)b * (LL + 4) + row) * DD + c] = (__bf16)0.f;
}

// -------------------------------------------------------------------------
// 4) conv1d as implicit GEMM with 2x2 register blocking:
//    y[b][l][co] = sum_t sum_ci x[b][l+t][ci] * wt[t][co][ci]
//    each wave: 32 rows x 32 cols; block (8 waves) = 64 rows x 128 cols.
// -------------------------------------------------------------------------
__global__ __launch_bounds__(256)
void conv_wmma(const __bf16* __restrict__ xin, const __bf16* __restrict__ wt,
               const float* __restrict__ bias, float* __restrict__ y,
               int rowsInPerB, int rowsOutPerB, int tiles32PerB, int Cout, int maskRows) {
  const int wave = threadIdx.x >> 5, lane = threadIdx.x & 31;
  const int mt = blockIdx.x * 2 + (wave >> 2);     // 32-row tile index
  const int b  = mt / tiles32PerB;
  const int l0 = (mt % tiles32PerB) * 32;
  const int n0 = (blockIdx.y * 4 + (wave & 3)) * 32;
  v8f c00 = {}, c01 = {}, c10 = {}, c11 = {};
  for (int t = 0; t < KK; t++) {
    const __bf16* pa = xin + ((size_t)b * rowsInPerB + l0 + t) * DD;
    const __bf16* pw = wt + ((size_t)t * Cout + n0) * DD;
    for (int k0 = 0; k0 < DD; k0 += 32) {
      v16bf a0 = load_frag(pa + k0, DD);
      v16bf a1 = load_frag(pa + (size_t)16 * DD + k0, DD);
      v16bf w0 = load_frag(pw + k0, DD);
      v16bf w1 = load_frag(pw + (size_t)16 * DD + k0, DD);
      c00 = wmma_bf16(a0, w0, c00);
      c01 = wmma_bf16(a0, w1, c01);
      c10 = wmma_bf16(a1, w0, c10);
      c11 = wmma_bf16(a1, w1, c11);
    }
  }
  const int rl = (lane >> 4) << 3, cn = lane & 15;
  const v8f accs[4] = {c00, c01, c10, c11};
  for (int q = 0; q < 4; q++) {
    const int mo = (q >> 1) * 16, col = n0 + (q & 1) * 16 + cn;
    const float bv = bias[col];
    for (int i = 0; i < 8; i++) {
      int l = l0 + mo + rl + i;
      if (l < maskRows)
        y[((size_t)b * rowsOutPerB + l) * Cout + col] = accs[q][i] + bv;
    }
  }
}

// -------------------------------------------------------------------------
// 5) BN statistics (training mode, biased var) per channel over (B, L)
// -------------------------------------------------------------------------
__global__ __launch_bounds__(256)
void bn_stats_kernel(const float* __restrict__ y, int C, int NL, float* stats) {
  const int c = blockIdx.x;
  const int wave = threadIdx.x >> 5, lane = threadIdx.x & 31;
  float s = 0.f, s2 = 0.f;
  for (int i = threadIdx.x; i < NL; i += 256) {
    float v = y[(size_t)i * C + c];
    s += v; s2 += v * v;
  }
  for (int o = 16; o; o >>= 1) { s += __shfl_xor(s, o); s2 += __shfl_xor(s2, o); }
  __shared__ float rs[8], rs2[8];
  if (lane == 0) { rs[wave] = s; rs2[wave] = s2; }
  __syncthreads();
  if (threadIdx.x == 0) {
    float a = 0.f, b2 = 0.f;
    for (int w = 0; w < 8; w++) { a += rs[w]; b2 += rs2[w]; }
    float mean = a / (float)NL;
    stats[c * 2] = mean;
    stats[c * 2 + 1] = b2 / (float)NL - mean * mean;
  }
}

// bn1 + relu + maxpool(k2 s2) -> zpad [B][1028][128] bf16 (rows>=1024 zero)
__global__ __launch_bounds__(256)
void bn1_pool_kernel(const float* __restrict__ y1, const float* __restrict__ st,
                     const float* g, const float* bb, __bf16* zpad) {
  int i = blockIdx.x * 256 + threadIdx.x;          // 16*1028*128
  if (i >= BB * (L2IN + 4) * C1) return;
  int c = i & 127, l2 = (i >> 7) % (L2IN + 4), b = i / ((L2IN + 4) * C1);
  float r = 0.f;
  if (l2 < L2IN) {
    float mean = st[c * 2], var = st[c * 2 + 1];
    float sc = g[c] * rsqrtf(var + EPS);
    float sh = bb[c] - mean * sc;
    float a0 = y1[((size_t)b * LL + l2 * 2) * C1 + c] * sc + sh;
    float a1 = y1[((size_t)b * LL + l2 * 2 + 1) * C1 + c] * sc + sh;
    r = fmaxf(fmaxf(a0, a1), 0.f);
  }
  zpad[((size_t)b * (L2IN + 4) + l2) * C1 + c] = (__bf16)r;
}

// bn2 + relu + global max over l -> m[b][c]
__global__ __launch_bounds__(256)
void bn2_gmax_kernel(const float* __restrict__ y2, const float* __restrict__ st,
                     const float* g, const float* bb, float* m) {
  int b = blockIdx.x, c = threadIdx.x;             // blockDim = 256 = C2
  float mean = st[c * 2], var = st[c * 2 + 1];
  float sc = g[c] * rsqrtf(var + EPS);
  float sh = bb[c] - mean * sc;
  float mx = 0.f;                                   // relu => max >= 0
  for (int l = 0; l < L2OUT; l++) {
    float v = y2[((size_t)b * L2OUT + l) * C2 + c] * sc + sh;
    mx = fmaxf(mx, fmaxf(v, 0.f));
  }
  m[b * C2 + c] = mx;
}

// tiny FC: out[b][nc] = m[b] . fc_w[nc] + fc_b[nc]
__global__ void fc_kernel(const float* m, const float* fw, const float* fb, float* out) {
  int t = threadIdx.x;
  if (t >= BB * 2) return;
  int b = t >> 1, nc = t & 1;
  float s = fb[nc];
  for (int c = 0; c < C2; c++) s += m[b * C2 + c] * fw[nc * C2 + c];
  out[b * 2 + nc] = s;
}

// -------------------------------------------------------------------------
// launch
// -------------------------------------------------------------------------
extern "C" void kernel_launch(void* const* d_in, const int* in_sizes, int n_in,
                              void* d_out, int out_size, void* d_ws, size_t ws_size,
                              hipStream_t stream) {
  const int*   X    = (const int*)d_in[0];
  const float* sa   = (const float*)d_in[1];
  const int*   ptm  = (const int*)d_in[2];
  const float* emb  = (const float*)d_in[3];
  const float* pemb = (const float*)d_in[4];
  const float* rpe  = (const float*)d_in[5];
  const float* ipw  = (const float*)d_in[6];
  const float* ipb  = (const float*)d_in[7];
  const float* opw  = (const float*)d_in[8];
  const float* opb  = (const float*)d_in[9];
  const float* l1w  = (const float*)d_in[10];
  const float* l1b  = (const float*)d_in[11];
  const float* l2w  = (const float*)d_in[12];
  const float* l2b  = (const float*)d_in[13];
  const float* ln1g = (const float*)d_in[14];
  const float* ln1b = (const float*)d_in[15];
  const float* ln2g = (const float*)d_in[16];
  const float* ln2b = (const float*)d_in[17];
  const float* c1w  = (const float*)d_in[18];
  const float* c1b  = (const float*)d_in[19];
  const float* bn1g = (const float*)d_in[20];
  const float* bn1b = (const float*)d_in[21];
  const float* c2w  = (const float*)d_in[22];
  const float* c2b  = (const float*)d_in[23];
  const float* bn2g = (const float*)d_in[24];
  const float* bn2b = (const float*)d_in[25];
  const float* fcw  = (const float*)d_in[26];
  const float* fcb  = (const float*)d_in[27];
  float* out = (float*)d_out;

  char* ws = (char*)d_ws;
  size_t off = 0;
  auto alloc = [&](size_t bytes) { size_t o = off; off = (off + bytes + 255) & ~(size_t)255; return o; };

  const size_t BL = (size_t)BB * LL;
  float*  x_f   = (float*) (ws + alloc(BL * DD * 4));
  __bf16* x_b   = (__bf16*)(ws + alloc(BL * DD * 2));
  __bf16* wq_b  = (__bf16*)(ws + alloc((size_t)3 * DD * DD * 2));
  __bf16* wo_b  = (__bf16*)(ws + alloc((size_t)DD * DD * 2));
  __bf16* w1_b  = (__bf16*)(ws + alloc((size_t)FF * DD * 2));
  __bf16* w2_b  = (__bf16*)(ws + alloc((size_t)DD * FF * 2));
  __bf16* w1t_b = (__bf16*)(ws + alloc((size_t)KK * C1 * DD * 2));
  __bf16* w2t_b = (__bf16*)(ws + alloc((size_t)KK * C2 * C1 * 2));
  __bf16* qkv_b = (__bf16*)(ws + alloc(BL * 3 * DD * 2));
  __bf16* vt_b  = (__bf16*)(ws + alloc((size_t)BB * HH * HD * LL * 2));
  __bf16* o_b   = (__bf16*)(ws + alloc(BL * DD * 2));
  float*  op_f  = (float*) (ws + alloc(BL * DD * 4));
  float*  x1_f  = (float*) (ws + alloc(BL * DD * 4));
  __bf16* x1_b  = (__bf16*)(ws + alloc(BL * DD * 2));
  __bf16* ff1_b = (__bf16*)(ws + alloc(BL * FF * 2));
  float*  ff2_f = (float*) (ws + alloc(BL * DD * 4));
  __bf16* xpad  = (__bf16*)(ws + alloc((size_t)BB * (LL + 4) * DD * 2));
  float*  y1_f  = (float*) (ws + alloc(BL * C1 * 4));
  float*  st1   = (float*) (ws + alloc(C1 * 2 * 4));
  __bf16* zpad  = (__bf16*)(ws + alloc((size_t)BB * (L2IN + 4) * C1 * 2));
  float*  y2_f  = (float*) (ws + alloc((size_t)BB * L2OUT * C2 * 4));
  float*  st2   = (float*) (ws + alloc(C2 * 2 * 4));
  float*  m_f   = (float*) (ws + alloc((size_t)BB * C2 * 4));
  (void)ws_size; (void)n_in; (void)in_sizes; (void)out_size;

  // 0) embedding + weight conversions
  embed_kernel<<<(BB * LL * DD + 255) / 256, 256, 0, stream>>>(X, sa, ptm, emb, pemb, x_f, x_b);
  f2bf_kernel<<<(3 * DD * DD + 255) / 256, 256, 0, stream>>>(ipw, wq_b, 3 * DD * DD);
  f2bf_kernel<<<(DD * DD + 255) / 256, 256, 0, stream>>>(opw, wo_b, DD * DD);
  f2bf_kernel<<<(FF * DD + 255) / 256, 256, 0, stream>>>(l1w, w1_b, FF * DD);
  f2bf_kernel<<<(DD * FF + 255) / 256, 256, 0, stream>>>(l2w, w2_b, DD * FF);
  convtr_kernel<<<(KK * C1 * DD + 255) / 256, 256, 0, stream>>>(c1w, w1t_b, C1, DD);
  convtr_kernel<<<(KK * C2 * C1 + 255) / 256, 256, 0, stream>>>(c2w, w2t_b, C2, C1);

  // 1) QKV projection (WMMA) -> bf16
  gemm_bf16<<<dim3(BL / 64, (3 * DD) / 128), 256, 0, stream>>>(
      x_b, wq_b, ipb, nullptr, qkv_b, (int)BL, 3 * DD, DD, 0);
  vtrans_kernel<<<(BB * HH * HD * LL + 255) / 256, 256, 0, stream>>>(qkv_b, vt_b);

  // 2) fused attention (WMMA + LDS softmax)
  attn_kernel<<<BB * HH * (LL / 16), 256, ATTN_SMEM, stream>>>(qkv_b, vt_b, rpe, o_b);

  // 3) output projection + residual + LN1
  gemm_bf16<<<dim3(BL / 64, DD / 128), 256, 0, stream>>>(
      o_b, wo_b, opb, op_f, nullptr, (int)BL, DD, DD, 0);
  add_ln_kernel<<<BL / 8, 256, 0, stream>>>(x_f, op_f, ln1g, ln1b, x1_f, x1_b, LL, 0);

  // 4) FFN (relu fused into lin1) + residual + LN2 (straight into padded conv input)
  gemm_bf16<<<dim3(BL / 64, FF / 128), 256, 0, stream>>>(
      x1_b, w1_b, l1b, nullptr, ff1_b, (int)BL, FF, DD, 1);
  gemm_bf16<<<dim3(BL / 64, DD / 128), 256, 0, stream>>>(
      ff1_b, w2_b, l2b, ff2_f, nullptr, (int)BL, DD, FF, 0);
  xpad_zero_kernel<<<(BB * 4 * DD + 255) / 256, 256, 0, stream>>>(xpad);
  add_ln_kernel<<<BL / 8, 256, 0, stream>>>(x1_f, ff2_f, ln2g, ln2b, nullptr, xpad, LL + 4, 2);

  // 5) conv1 (implicit GEMM, WMMA) -> BN1 stats -> bn+relu+pool
  conv_wmma<<<dim3(BB * (LL / 32) / 2, C1 / 128), 256, 0, stream>>>(
      xpad, w1t_b, c1b, y1_f, LL + 4, LL, LL / 32, C1, LL);
  bn_stats_kernel<<<C1, 256, 0, stream>>>(y1_f, C1, BB * LL, st1);
  bn1_pool_kernel<<<(BB * (L2IN + 4) * C1 + 255) / 256, 256, 0, stream>>>(
      y1_f, st1, bn1g, bn1b, zpad);

  // 6) conv2 (valid, WMMA) -> BN2 stats -> bn+relu+global-max -> FC
  conv_wmma<<<dim3(BB * (L2IN / 32) / 2, C2 / 128), 256, 0, stream>>>(
      zpad, w2t_b, c2b, y2_f, L2IN + 4, L2OUT, L2IN / 32, C2, L2OUT);
  bn_stats_kernel<<<C2, 256, 0, stream>>>(y2_f, C2, BB * L2OUT, st2);
  bn2_gmax_kernel<<<BB, C2, 0, stream>>>(y2_f, st2, bn2g, bn2b, m_f);
  fc_kernel<<<1, 64, 0, stream>>>(m_f, fcw, fcb, out);
}
